// Modular_GCN_41094247088195
// MI455X (gfx1250) — matrix operand order repaired and checked by reading, not compiled, not run on hardware
//
#include <hip/hip_runtime.h>
#include <hip/hip_bf16.h>
#include <stdint.h>

// ---------------- WMMA types (CDNA5 / gfx1250, wave32) ----------------
typedef __attribute__((ext_vector_type(16))) __bf16 v16bf;
typedef __attribute__((ext_vector_type(8)))  float  v8f;
typedef __attribute__((ext_vector_type(4)))  unsigned u32x4;

struct U8x32 { u32x4 lo; u32x4 hi; };   // 32 bytes -> bit_cast to v16bf

// order-preserving float <-> uint map for atomicMax-based segment max
__device__ __forceinline__ unsigned ford(float f) {
    unsigned u = __float_as_uint(f);
    return (u & 0x80000000u) ? ~u : (u | 0x80000000u);
}
__device__ __forceinline__ float unford(unsigned o) {
    unsigned u = (o & 0x80000000u) ? (o & 0x7FFFFFFFu) : ~o;
    return __uint_as_float(u);
}

__device__ __forceinline__ float leaky(float v, float s) { return v > 0.0f ? v : s * v; }

__device__ __forceinline__ unsigned short f2bf(float f) {
    unsigned u = __float_as_uint(f);
    u += 0x7FFFu + ((u >> 16) & 1u);
    return (unsigned short)(u >> 16);
}

// ---------------- fp32 -> bf16 (row-major copy, for A) ----------------
__global__ void k_f32_to_bf16(const float* __restrict__ in,
                              unsigned short* __restrict__ out, long n) {
    long i = (long)blockIdx.x * blockDim.x + threadIdx.x;
    if (i < n) out[i] = f2bf(in[i]);
}

// ---------------- weight swizzle: W[K x Mo] f32 -> per-lane fragment-major bf16 ----
// Bs index = ((kt*tilesN + ct)*32 + lane)*16 + i
// lane L holds col = ct*16 + (L&15); element i -> K = kt*32 + (L>>4)*16 + i
__global__ void k_swizzle_B(const float* __restrict__ W,
                            unsigned short* __restrict__ Bs, int K, int Mo) {
    int tilesN = Mo >> 4;
    long tot = (long)K * Mo;
    long idx = (long)blockIdx.x * blockDim.x + threadIdx.x;
    if (idx >= tot) return;
    int i = (int)(idx & 15);
    int L = (int)((idx >> 4) & 31);
    long rest = idx >> 9;
    int ct = (int)(rest % tilesN);
    int kt = (int)(rest / tilesN);
    int k   = kt * 32 + (L >> 4) * 16 + i;
    int col = ct * 16 + (L & 15);
    Bs[idx] = f2bf(W[(size_t)k * Mo + col]);
}

// ---------------- GEMM via v_wmma_f32_16x16x32_bf16 ----------------
// A: n x 128 bf16 row-major; Bs: swizzled fragments; C: n x MO f32 = A@B + bias
// 256 threads = 8 waves per block, one 16x16 tile per wave, K unrolled (4 WMMA).
template<int MO>
__global__ __launch_bounds__(256)
void k_gemm_wmma(const unsigned short* __restrict__ A,
                 const unsigned short* __restrict__ Bs,
                 const float* __restrict__ bias,
                 float* __restrict__ C, int n) {
    constexpr int K  = 128;
    constexpr int TN = MO / 16;
    const int wave = threadIdx.x >> 5;
    const int lane = threadIdx.x & 31;
    const int tilesM = (n + 15) >> 4;
    const int tile = blockIdx.x * 8 + wave;
    if (tile >= tilesM * TN) return;
    const int tm = tile / TN, tn = tile % TN;
    const int m = lane & 15, g = lane >> 4;
    int row = tm * 16 + m;
    if (row >= n) row = n - 1;                       // clamp loads only
    const int coln = tn * 16 + m;

    v8f acc = {};
    const unsigned short* arow = A + (size_t)row * K;
#pragma unroll
    for (int kt = 0; kt < K / 32; kt++) {
        const int k0 = kt * 32;
        U8x32 ua, ub;
        ua.lo = *(const u32x4*)(arow + k0 + g * 16);       // K = k0+16g .. +7
        ua.hi = *(const u32x4*)(arow + k0 + g * 16 + 8);   // K = k0+16g+8 .. +15
        const unsigned short* bp = Bs + ((size_t)(kt * TN + tn) * 32 + lane) * 16;
        ub.lo = *(const u32x4*)(bp);
        ub.hi = *(const u32x4*)(bp + 8);
        v16bf a = __builtin_bit_cast(v16bf, ua);
        v16bf b = __builtin_bit_cast(v16bf, ub);
        acc = __builtin_amdgcn_wmma_f32_16x16x32_bf16(
            false, a, false, b, (short)0, acc, false, false);
    }

    const float bv = bias[coln];
    if (tm * 16 + 16 <= n) {                         // full tile: immediate offsets
        float* cp = C + (size_t)(tm * 16 + 8 * g) * MO + coln;
#pragma unroll
        for (int r = 0; r < 8; r++) cp[r * MO] = acc[r] + bv;
    } else {
#pragma unroll
        for (int r = 0; r < 8; r++) {
            int rr = tm * 16 + r + 8 * g;
            if (rr < n) C[(size_t)rr * MO + coln] = acc[r] + bv;
        }
    }
}

// ---------------- edge pass 1: attention logits + segment max ----------------
__global__ void k_edge_logits(const int* __restrict__ ei, int E, int n,
                              const float* __restrict__ xl, const float* __restrict__ xr,
                              const float* __restrict__ att, int H, int C,
                              float* __restrict__ eb, unsigned* __restrict__ mord) {
    long t = (long)blockIdx.x * blockDim.x + threadIdx.x;
    long tot = (long)(E + n) * H;
    if (t >= tot) return;
    int e = (int)(t / H), h = (int)(t % H);
    int src = (e < E) ? ei[e]     : (e - E);
    int dst = (e < E) ? ei[E + e] : (e - E);
    int HC = H * C;
    const float4* pl = (const float4*)(xl + (size_t)src * HC + h * C);
    const float4* pr = (const float4*)(xr + (size_t)dst * HC + h * C);
    const float4* pa = (const float4*)(att + h * C);
    float s = 0.0f;
    for (int c = 0; c < C / 4; c++) {
        float4 l = pl[c], r = pr[c], a = pa[c];
        s += leaky(l.x + r.x, 0.2f) * a.x + leaky(l.y + r.y, 0.2f) * a.y
           + leaky(l.z + r.z, 0.2f) * a.z + leaky(l.w + r.w, 0.2f) * a.w;
    }
    eb[t] = s;
    atomicMax(&mord[(size_t)dst * H + h], ford(s));
}

// ---------------- edge pass 2: exp, segment sum, weighted scatter ----------------
__global__ void k_edge_accum(const int* __restrict__ ei, int E, int n,
                             const float* __restrict__ xl,
                             const float* __restrict__ eb,
                             const unsigned* __restrict__ mord, int H, int C,
                             float* __restrict__ ssum, float* __restrict__ acc) {
    long t = (long)blockIdx.x * blockDim.x + threadIdx.x;
    long tot = (long)(E + n) * H;
    if (t >= tot) return;
    int e = (int)(t / H), h = (int)(t % H);
    int src = (e < E) ? ei[e]     : (e - E);
    int dst = (e < E) ? ei[E + e] : (e - E);
    int HC = H * C;
    float mx = unford(mord[(size_t)dst * H + h]);
    float ex = __expf(eb[t] - mx);
    atomicAdd(&ssum[(size_t)dst * H + h], ex);
    const float4* pl = (const float4*)(xl + (size_t)src * HC + h * C);
    float* pa = acc + (size_t)dst * HC + h * C;
    for (int c = 0; c < C / 4; c++) {
        float4 v = pl[c];
        atomicAdd(pa + 4 * c + 0, ex * v.x);
        atomicAdd(pa + 4 * c + 1, ex * v.y);
        atomicAdd(pa + 4 * c + 2, ex * v.z);
        atomicAdd(pa + 4 * c + 3, ex * v.w);
    }
}

// ---------------- node finalize: normalize softmax + GAT bias ----------------
__global__ void k_node_finalize(const float* __restrict__ acc,
                                const float* __restrict__ ssum,
                                const float* __restrict__ gb,
                                int n, int H, int C, float* __restrict__ out) {
    long idx = (long)blockIdx.x * blockDim.x + threadIdx.x;
    int HC = H * C;
    long tot = (long)n * HC;
    if (idx >= tot) return;
    int node = (int)(idx / HC);
    int f = (int)(idx % HC);
    int h = f / C;
    out[idx] = acc[idx] / ssum[(size_t)node * H + h] + gb[f];
}

// ---------------- BatchNorm column stats (sum, sum^2) ----------------
__global__ void k_col_stats(const float* __restrict__ x, int n, int F,
                            float* __restrict__ csum, float* __restrict__ csq) {
    int f = threadIdx.x;                   // blockDim.x == F
    int r0 = blockIdx.x * 256;
    int rend = r0 + 256; if (rend > n) rend = n;
    float s = 0.0f, q = 0.0f;
    for (int r = r0; r < rend; r++) {
        float v = x[(size_t)r * F + f];
        s += v; q += v * v;
    }
    atomicAdd(&csum[f], s);
    atomicAdd(&csq[f], q);
}

// ---------------- BatchNorm apply + LeakyReLU(0.01) (in place) ----------------
__global__ void k_bn_apply(float* __restrict__ x, int n, int F,
                           const float* __restrict__ csum, const float* __restrict__ csq,
                           const float* __restrict__ g, const float* __restrict__ b) {
    long idx = (long)blockIdx.x * blockDim.x + threadIdx.x;
    long tot = (long)n * F;
    if (idx >= tot) return;
    int f = (int)(idx % F);
    float inv_n = 1.0f / (float)n;
    float mu = csum[f] * inv_n;
    float var = csq[f] * inv_n - mu * mu;
    float v = (x[idx] - mu) * rsqrtf(var + 1e-5f) * g[f] + b[f];
    x[idx] = leaky(v, 0.01f);
}

// ---------------- global mean pool (atomics into [G,16]) ----------------
__global__ void k_pool(const float* __restrict__ x, const int* __restrict__ batch,
                       int n, int F, float* __restrict__ psum, float* __restrict__ pcnt) {
    long idx = (long)blockIdx.x * blockDim.x + threadIdx.x;
    long tot = (long)n * F;
    if (idx >= tot) return;
    int node = (int)(idx / F);
    int f = (int)(idx % F);
    int g = batch[node];
    atomicAdd(&psum[(size_t)g * F + f], x[idx]);
    if (f == 0) atomicAdd(&pcnt[g], 1.0f);
}

// ---------------- dense head: 16 -> 32 -> 16, one thread per graph ----------------
__global__ void k_head(const float* __restrict__ psum, const float* __restrict__ pcnt,
                       const float* __restrict__ W1, const float* __restrict__ b1,
                       const float* __restrict__ W2, const float* __restrict__ b2,
                       float* __restrict__ out, int G) {
    int g = threadIdx.x;
    if (g >= G) return;
    float cnt = pcnt[g] > 1.0f ? pcnt[g] : 1.0f;
    float pooled[16];
    for (int i = 0; i < 16; i++) pooled[i] = psum[g * 16 + i] / cnt;
    float h1[32];
    for (int j = 0; j < 32; j++) {
        float s = b1[j];
        for (int i = 0; i < 16; i++) s += pooled[i] * W1[i * 32 + j];
        h1[j] = leaky(s, 0.01f);
    }
    for (int j = 0; j < 16; j++) {
        float s = b2[j];
        for (int i = 0; i < 32; i++) s += h1[i] * W2[i * 16 + j];
        out[g * 16 + j] = leaky(s, 0.01f);
    }
}

// =====================================================================
extern "C" void kernel_launch(void* const* d_in, const int* in_sizes, int n_in,
                              void* d_out, int out_size, void* d_ws, size_t ws_size,
                              hipStream_t stream) {
    const int F_IN = 128, G = 64;
    const float* x0    = (const float*)d_in[0];
    const int*   ei    = (const int*)d_in[1];
    const int*   batch = (const int*)d_in[2];
    const int N = in_sizes[0] / F_IN;
    const int E = in_sizes[1] / 2;

    // ---- workspace carve (256B aligned) ----
    char* ws = (char*)d_ws;
    size_t off = 0;
    auto carve = [&](size_t bytes) -> void* {
        void* p = ws + off;
        off += (bytes + 255) & ~(size_t)255;
        return p;
    };
    unsigned short* xb   = (unsigned short*)carve((size_t)N * 128 * 2);
    unsigned short* wlb  = (unsigned short*)carve(128 * 128 * 2);
    unsigned short* wrb  = (unsigned short*)carve(128 * 128 * 2);
    float*    xl   = (float*)carve((size_t)N * 128 * 4);
    float*    xr   = (float*)carve((size_t)N * 128 * 4);
    float*    acc  = (float*)carve((size_t)N * 128 * 4);
    float*    xcur = (float*)carve((size_t)N * 128 * 4);
    float*    eb   = (float*)carve((size_t)(E + N) * 4 * 4);
    unsigned* mord = (unsigned*)carve((size_t)N * 4 * 4);
    float*    ssum = (float*)carve((size_t)N * 4 * 4);
    float*    csum = (float*)carve(128 * 4);
    float*    csq  = (float*)carve(128 * 4);
    float*    psum = (float*)carve((size_t)G * 16 * 4);
    float*    pcnt = (float*)carve((size_t)G * 4);

    struct LayerP { int iWl, ibl, iWr, ibr, iatt, igb, ibng, ibnb, H, C; };
    const LayerP LP[3] = {
        { 3,  4,  5,  6,  7,  8, 21, 22, 4, 32 },
        { 9, 10, 11, 12, 13, 14, 23, 24, 4, 32 },
        {15, 16, 17, 18, 19, 20, 25, 26, 1, 16 },
    };

    const int T = 256;
    const float* xin = x0;

    for (int l = 0; l < 3; l++) {
        const LayerP& p = LP[l];
        const int H = p.H, C = p.C, Fin = F_IN, Fout = H * C;
        const float* Wl  = (const float*)d_in[p.iWl];
        const float* bl  = (const float*)d_in[p.ibl];
        const float* Wr  = (const float*)d_in[p.iWr];
        const float* br  = (const float*)d_in[p.ibr];
        const float* att = (const float*)d_in[p.iatt];
        const float* gb  = (const float*)d_in[p.igb];
        const float* bng = (const float*)d_in[p.ibng];
        const float* bnb = (const float*)d_in[p.ibnb];

        // bf16 conversions (A row-major, B swizzled into fragment-major)
        long nx = (long)N * Fin;
        k_f32_to_bf16<<<(unsigned)((nx + T - 1) / T), T, 0, stream>>>(xin, xb, nx);
        long nw = (long)Fin * Fout;
        k_swizzle_B<<<(unsigned)((nw + T - 1) / T), T, 0, stream>>>(Wl, wlb, Fin, Fout);
        k_swizzle_B<<<(unsigned)((nw + T - 1) / T), T, 0, stream>>>(Wr, wrb, Fin, Fout);

        // xl = x@Wl + bl ; xr = x@Wr + br  (WMMA bf16 -> f32)
        int tilesM = (N + 15) / 16;
        int totTiles = tilesM * (Fout / 16);
        unsigned gblk = (unsigned)((totTiles + 7) / 8);
        if (Fout == 128) {
            k_gemm_wmma<128><<<gblk, 256, 0, stream>>>(xb, wlb, bl, xl, N);
            k_gemm_wmma<128><<<gblk, 256, 0, stream>>>(xb, wrb, br, xr, N);
        } else {
            k_gemm_wmma<16><<<gblk, 256, 0, stream>>>(xb, wlb, bl, xl, N);
            k_gemm_wmma<16><<<gblk, 256, 0, stream>>>(xb, wrb, br, xr, N);
        }

        // zero segment buffers (memset 0 == ordered -inf sentinel for max)
        hipMemsetAsync(mord, 0, (size_t)N * H * 4, stream);
        hipMemsetAsync(ssum, 0, (size_t)N * H * 4, stream);
        hipMemsetAsync(acc,  0, (size_t)N * Fout * 4, stream);

        long totE = (long)(E + N) * H;
        k_edge_logits<<<(unsigned)((totE + T - 1) / T), T, 0, stream>>>(
            ei, E, N, xl, xr, att, H, C, eb, mord);
        k_edge_accum<<<(unsigned)((totE + T - 1) / T), T, 0, stream>>>(
            ei, E, N, xl, eb, mord, H, C, ssum, acc);

        long totN = (long)N * Fout;
        k_node_finalize<<<(unsigned)((totN + T - 1) / T), T, 0, stream>>>(
            acc, ssum, gb, N, H, C, xcur);

        // BatchNorm (+ LeakyReLU 0.01)
        hipMemsetAsync(csum, 0, Fout * 4, stream);
        hipMemsetAsync(csq,  0, Fout * 4, stream);
        k_col_stats<<<(unsigned)((N + 255) / 256), Fout, 0, stream>>>(xcur, N, Fout, csum, csq);
        k_bn_apply<<<(unsigned)((totN + T - 1) / T), T, 0, stream>>>(
            xcur, N, Fout, csum, csq, bng, bnb);

        xin = xcur;
    }

    // global mean pool + dense head
    hipMemsetAsync(psum, 0, (size_t)G * 16 * 4, stream);
    hipMemsetAsync(pcnt, 0, (size_t)G * 4, stream);
    long totP = (long)N * 16;
    k_pool<<<(unsigned)((totP + T - 1) / T), T, 0, stream>>>(xcur, batch, N, 16, psum, pcnt);
    k_head<<<1, G, 0, stream>>>(psum, pcnt,
                                (const float*)d_in[27], (const float*)d_in[28],
                                (const float*)d_in[29], (const float*)d_in[30],
                                (float*)d_out, G);
}
